// NaiveSwitchTransformerEncoderLayer_31739808317984
// MI455X (gfx1250) — matrix operand order repaired and checked by reading, compiled, not run on hardware
//
#include <hip/hip_runtime.h>
#include <hip/hip_bf16.h>

// ---------------------------------------------------------------------------
// Switch-Transformer encoder layer for gfx1250 (MI455X).
// wave32 + v_wmma_f32_16x16x32_bf16 + double-buffered
// global_load_async_to_lds_b128 panel staging (ASYNCcnt pipelining).
// ---------------------------------------------------------------------------

#define BB 4
#define SS 1024
#define DD 1024
#define HH 16
#define HD 64
#define EE 8
#define FF 4096
#define TT (BB*SS)          // 4096 tokens
#define N3 (3*DD)           // 3072
#define MAXG ((TT/64)+EE)   // 72 expert-uniform 64-token groups max
#define PERMCAP (MAXG*64)   // 4608

typedef __attribute__((ext_vector_type(16))) __bf16 v16bf;
typedef __attribute__((ext_vector_type(8)))  float  v8f;

__device__ __forceinline__ v8f wmma_bf16(v16bf a, v16bf b, v8f c) {
  return __builtin_amdgcn_wmma_f32_16x16x32_bf16(false, a, false, b, (short)0, c, false, false);
}

__device__ __forceinline__ void setpair(v16bf& f, int v, unsigned u) {
  union { unsigned u; __bf16 h[2]; } cv; cv.u = u;
  f[2*v]   = cv.h[0];
  f[2*v+1] = cv.h[1];
}

// A-fragment K offset (16-bit A 16x32 layout, per ISA 7.12.2)
__device__ __forceinline__ int a_koff(int v, int half) {
  return (v < 4) ? (half*8 + 2*v) : (16 + half*8 + 2*(v-4));
}
// B-fragment K offset (K x 16 layout, same K order as A)
__device__ __forceinline__ int b_koff(int v, int half) {
  return (v < 4) ? (half*8 + 2*v) : (16 + half*8 + 2*(v-4));
}

// CDNA5 async global->LDS copy (16B per lane), tracked by ASYNCcnt.
__device__ __forceinline__ void async_load16(void* lds_ptr, const void* gptr) {
  unsigned lds_off = (unsigned)(unsigned long long)lds_ptr;
  asm volatile("global_load_async_to_lds_b128 %0, %1, off"
               :: "v"(lds_off), "v"((unsigned long long)gptr) : "memory");
}
__device__ __forceinline__ void wait_async() {
  asm volatile("s_wait_asynccnt 0" ::: "memory");
}

// ---------------------------------------------------------------------------
// Dense tiled GEMM: C[M,N] = A[M,K] @ W^T + bias, W stored [N,K] (K contig).
// 128 threads (4 waves), C block 128x64, KSTEP 64, double-buffered panels.
// Wave w owns cols [n0+16w, n0+16w+16), 8 M-tiles.
// ---------------------------------------------------------------------------
template<bool A_BF16>
__global__ __launch_bounds__(128)
void gemm_dense_tiled(const void* __restrict__ A_,
                      const float* __restrict__ W,
                      const float* __restrict__ bias,
                      float* __restrict__ Cf,
                      __bf16* __restrict__ Cb,
                      int M, int N, int K) {
  __shared__ __align__(16) char AsRaw[2][128 * 64 * 4];
  __shared__ __align__(16) float Bs[2][64 * 64];     // [col][k]

  const int tid = threadIdx.x;
  const int wave = tid >> 5, lane = tid & 31;
  const int half = lane >> 4, lr = lane & 15;
  const int m0 = blockIdx.x * 128;
  const int n0 = blockIdx.y * 64;
  const int colw = n0 + wave * 16 + lr;

  v8f acc[8] = {};

  auto stage = [&](int kb, int bi) {
    if (A_BF16) {
      const __bf16* Ab = (const __bf16*)A_;
      __bf16* Asb = (__bf16*)AsRaw[bi];
#pragma unroll
      for (int i = 0; i < 8; ++i) {            // 1024 x 16B chunks
        const int ci = tid + 128 * i;
        const int row = ci >> 3, kc = (ci & 7) * 8;
        async_load16(&Asb[row * 64 + kc], Ab + (size_t)(m0 + row) * K + kb + kc);
      }
    } else {
      const float* Af = (const float*)A_;
      float* Asf = (float*)AsRaw[bi];
#pragma unroll
      for (int i = 0; i < 16; ++i) {           // 2048 x 16B chunks
        const int ci = tid + 128 * i;
        const int row = ci >> 4, kc = (ci & 15) * 4;
        async_load16(&Asf[row * 64 + kc], Af + (size_t)(m0 + row) * K + kb + kc);
      }
    }
#pragma unroll
    for (int i = 0; i < 8; ++i) {              // 1024 x 16B chunks
      const int ci = tid + 128 * i;
      const int col = ci >> 4, kc = (ci & 15) * 4;
      async_load16(&Bs[bi][col * 64 + kc], W + (size_t)(n0 + col) * K + kb + kc);
    }
  };

  stage(0, 0);
  wait_async();
  __syncthreads();

  const int nk = K >> 6;
  for (int it = 0; it < nk; ++it) {
    const int cur = it & 1;
    if (it + 1 < nk) stage((it + 1) << 6, cur ^ 1);   // prefetch next panel

    const float*  Asf = (const float*) AsRaw[cur];
    const __bf16* Asb = (const __bf16*)AsRaw[cur];
    const float*  Bsc = Bs[cur];

#pragma unroll
    for (int ks = 0; ks < 64; ks += 32) {
      v16bf bfrag;
#pragma unroll
      for (int v = 0; v < 8; ++v) {
        const int kk = ks + b_koff(v, half);
        float2 w2 = *(const float2*)&Bsc[(wave * 16 + lr) * 64 + kk];
        bfrag[2*v] = (__bf16)w2.x; bfrag[2*v+1] = (__bf16)w2.y;
      }
#pragma unroll
      for (int mt = 0; mt < 8; ++mt) {
        v16bf afrag;
#pragma unroll
        for (int v = 0; v < 8; ++v) {
          const int ka = ks + a_koff(v, half);
          if (A_BF16) {
            setpair(afrag, v, *(const unsigned*)&Asb[(mt * 16 + lr) * 64 + ka]);
          } else {
            float2 f2 = *(const float2*)&Asf[(mt * 16 + lr) * 64 + ka];
            afrag[2*v] = (__bf16)f2.x; afrag[2*v+1] = (__bf16)f2.y;
          }
        }
        acc[mt] = wmma_bf16(afrag, bfrag, acc[mt]);
      }
    }
    wait_async();
    __syncthreads();
  }

  const float bval = bias ? bias[colw] : 0.0f;
#pragma unroll
  for (int mt = 0; mt < 8; ++mt) {
#pragma unroll
    for (int r = 0; r < 8; ++r) {
      const int rowc = m0 + mt * 16 + r + half * 8;
      const float val = acc[mt][r] + bval;
      if (Cf) Cf[(size_t)rowc * N + colw] = val;
      else    Cb[(size_t)rowc * N + colw] = (__bf16)val;
    }
  }
}

// ---------------------------------------------------------------------------
// Flash attention: 128-thread block = 4 waves; wave w owns query tile
// (blockIdx.x*4 + w) of one (b,h). K/V tiles async-staged into LDS shared by
// all 4 waves, double-buffered. qkv bf16 [T,3D]: Q +0, K +1024, V +2048.
// ---------------------------------------------------------------------------
__global__ __launch_bounds__(128)
void attn_kernel(const __bf16* __restrict__ qkv,
                 __bf16* __restrict__ attn_out) {
  const int tid = threadIdx.x;
  const int wv = tid >> 5, lane = tid & 31;
  const int half = lane >> 4, lr = lane & 15;
  const int hidx = blockIdx.y, bidx = blockIdx.z;
  const int qt = blockIdx.x * 4 + wv;
  const int q0 = bidx * SS + qt * 16;

  __shared__ __align__(16) __bf16 kbuf[2][32 * 64];   // [key][dim]
  __shared__ __align__(16) __bf16 vbuf[2][32 * 64];   // [key][dim]
  __shared__ __align__(16) __bf16 pbuf[4][16 * 32];   // per-wave P scratch

  v16bf qa0, qa1;
  {
    const __bf16* qrow = qkv + (size_t)(q0 + lr) * N3 + hidx * HD;
#pragma unroll
    for (int v = 0; v < 8; ++v) {
      const int ka = a_koff(v, half);
      setpair(qa0, v, *(const unsigned*)(qrow + ka));
      setpair(qa1, v, *(const unsigned*)(qrow + 32 + ka));
    }
  }

  v8f o0 = {}, o1 = {}, o2 = {}, o3 = {};
  float m[8], l[8];
#pragma unroll
  for (int r = 0; r < 8; ++r) { m[r] = -1e30f; l[r] = 0.0f; }

  auto stage = [&](int kb, int bi) {
#pragma unroll
    for (int i = 0; i < 2; ++i) {
      const int ci = tid + 128 * i;
      const int key = ci >> 3, c8 = (ci & 7) * 8;
      const size_t tok = (size_t)(bidx * SS + kb + key) * N3 + hidx * HD + c8;
      async_load16(&kbuf[bi][key * 64 + c8], qkv + tok + 1024);
      async_load16(&vbuf[bi][key * 64 + c8], qkv + tok + 2048);
    }
  };

  stage(0, 0);
  wait_async();
  __syncthreads();

  for (int it = 0; it < SS / 32; ++it) {
    const int cur = it & 1;
    if (it + 1 < SS / 32) stage((it + 1) * 32, cur ^ 1);

    v8f sa = {}, sb = {};
#pragma unroll
    for (int sub = 0; sub < 2; ++sub) {
      v16bf bk0, bk1;
#pragma unroll
      for (int v = 0; v < 8; ++v) {
        const int kk = b_koff(v, half);
        setpair(bk0, v, *(const unsigned*)&kbuf[cur][(sub * 16 + lr) * 64 + kk]);
        setpair(bk1, v, *(const unsigned*)&kbuf[cur][(sub * 16 + lr) * 64 + 32 + kk]);
      }
      if (sub == 0) { sa = wmma_bf16(qa0, bk0, sa); sa = wmma_bf16(qa1, bk1, sa); }
      else          { sb = wmma_bf16(qa0, bk0, sb); sb = wmma_bf16(qa1, bk1, sb); }
    }

#pragma unroll
    for (int r = 0; r < 8; ++r) {
      float xa = sa[r] * 0.125f;
      float xb = sb[r] * 0.125f;
      float mx = fmaxf(xa, xb);
      mx = fmaxf(mx, __shfl_xor(mx, 1, 16));
      mx = fmaxf(mx, __shfl_xor(mx, 2, 16));
      mx = fmaxf(mx, __shfl_xor(mx, 4, 16));
      mx = fmaxf(mx, __shfl_xor(mx, 8, 16));
      const float mnew = fmaxf(m[r], mx);
      const float corr = __expf(m[r] - mnew);
      const float pa = __expf(xa - mnew);
      const float pb = __expf(xb - mnew);
      float ps = pa + pb;
      ps += __shfl_xor(ps, 1, 16);
      ps += __shfl_xor(ps, 2, 16);
      ps += __shfl_xor(ps, 4, 16);
      ps += __shfl_xor(ps, 8, 16);
      l[r] = l[r] * corr + ps;
      m[r] = mnew;
      o0[r] *= corr; o1[r] *= corr; o2[r] *= corr; o3[r] *= corr;
      pbuf[wv][(r + half * 8) * 32 +      lr] = (__bf16)pa;
      pbuf[wv][(r + half * 8) * 32 + 16 + lr] = (__bf16)pb;
    }

    v16bf pfrag;
#pragma unroll
    for (int v = 0; v < 8; ++v) {
      const int ka = a_koff(v, half);
      setpair(pfrag, v, *(const unsigned*)(&pbuf[wv][lr * 32 + ka]));
    }
#pragma unroll
    for (int nt = 0; nt < 4; ++nt) {
      v16bf bv;
#pragma unroll
      for (int v = 0; v < 8; ++v) {
        const int kk = b_koff(v, half);
        bv[2*v]   = vbuf[cur][kk * 64 + nt * 16 + lr];
        bv[2*v+1] = vbuf[cur][(kk + 1) * 64 + nt * 16 + lr];
      }
      if      (nt == 0) o0 = wmma_bf16(pfrag, bv, o0);
      else if (nt == 1) o1 = wmma_bf16(pfrag, bv, o1);
      else if (nt == 2) o2 = wmma_bf16(pfrag, bv, o2);
      else              o3 = wmma_bf16(pfrag, bv, o3);
    }
    wait_async();
    __syncthreads();
  }

#pragma unroll
  for (int r = 0; r < 8; ++r) {
    const float inv = 1.0f / l[r];
    __bf16* orow = attn_out + (size_t)(q0 + r + half * 8) * DD + hidx * HD;
    orow[ 0 + lr] = (__bf16)(o0[r] * inv);
    orow[16 + lr] = (__bf16)(o1[r] * inv);
    orow[32 + lr] = (__bf16)(o2[r] * inv);
    orow[48 + lr] = (__bf16)(o3[r] * inv);
  }
}

// ---------------------------------------------------------------------------
// out = LayerNorm(a + b) * g + beta ; one row per block
// ---------------------------------------------------------------------------
__global__ void add_ln_kernel(const float* __restrict__ a,
                              const float* __restrict__ b,
                              const float* __restrict__ g,
                              const float* __restrict__ beta,
                              float* __restrict__ out) {
  const int rowi = blockIdx.x;
  const int tid = threadIdx.x;
  __shared__ float buf[DD];
  __shared__ float red[8];
  __shared__ float stats[2];

  const float* ar = a + (size_t)rowi * DD;
  const float* br = b + (size_t)rowi * DD;

  float s = 0.0f;
  for (int d = tid; d < DD; d += 256) {
    const float v = ar[d] + br[d];
    buf[d] = v;
    s += v;
  }
  for (int off = 16; off >= 1; off >>= 1) s += __shfl_xor(s, off);
  if ((tid & 31) == 0) red[tid >> 5] = s;
  __syncthreads();
  if (tid == 0) {
    float t = 0.0f;
    for (int i = 0; i < 8; ++i) t += red[i];
    stats[0] = t * (1.0f / DD);
  }
  __syncthreads();
  const float mean = stats[0];

  float vs = 0.0f;
  for (int d = tid; d < DD; d += 256) {
    const float dd = buf[d] - mean;
    vs += dd * dd;
  }
  for (int off = 16; off >= 1; off >>= 1) vs += __shfl_xor(vs, off);
  if ((tid & 31) == 0) red[tid >> 5] = vs;
  __syncthreads();
  if (tid == 0) {
    float t = 0.0f;
    for (int i = 0; i < 8; ++i) t += red[i];
    stats[1] = t * (1.0f / DD);
  }
  __syncthreads();
  const float rinv = rsqrtf(stats[1] + 1e-5f);
  for (int d = tid; d < DD; d += 256)
    out[(size_t)rowi * DD + d] = (buf[d] - mean) * rinv * g[d] + beta[d];
}

// ---------------------------------------------------------------------------
// Routing (expert segments padded to 64 so every 64-row group is uniform)
// ---------------------------------------------------------------------------
__global__ void gate_kernel(const float* __restrict__ y1,
                            const float* __restrict__ gw,
                            const float* __restrict__ gb,
                            int* __restrict__ expert_of,
                            int* __restrict__ counts) {
  const int lane = threadIdx.x;
  const int t = blockIdx.x * blockDim.y + threadIdx.y;
  if (t >= TT) return;
  const float* row = y1 + (size_t)t * DD;
  float acc[EE];
#pragma unroll
  for (int e = 0; e < EE; ++e) acc[e] = 0.0f;
  for (int d = lane; d < DD; d += 32) {
    const float xv = row[d];
#pragma unroll
    for (int e = 0; e < EE; ++e) acc[e] += xv * gw[e * DD + d];
  }
#pragma unroll
  for (int e = 0; e < EE; ++e)
    for (int off = 16; off >= 1; off >>= 1) acc[e] += __shfl_xor(acc[e], off);
  if (lane == 0) {
    int best = 0;
    float bv = acc[0] + gb[0];
#pragma unroll
    for (int e = 1; e < EE; ++e) {
      const float v = acc[e] + gb[e];
      if (v > bv) { bv = v; best = e; }
    }
    expert_of[t] = best;
    atomicAdd(&counts[best], 1);
  }
}

__global__ void route_init_kernel(int* perm, int* counts, int* cursor) {
  const int i = blockIdx.x * 256 + threadIdx.x;
  if (i < PERMCAP) perm[i] = -1;
  if (i < EE) { counts[i] = 0; cursor[i] = 0; }
}

__global__ void route_offsets_kernel(const int* __restrict__ counts,
                                     int* __restrict__ offsets,
                                     int* __restrict__ group_expert) {
  if (threadIdx.x == 0 && blockIdx.x == 0) {
    int off = 0;
    for (int e = 0; e < EE; ++e) {
      offsets[e] = off;
      off += (counts[e] + 63) & ~63;
    }
    offsets[EE] = off;
    for (int e = 0; e < EE; ++e) {
      const int ng = ((counts[e] + 63) & ~63) >> 6;
      for (int j = 0; j < ng; ++j) group_expert[(offsets[e] >> 6) + j] = e;
    }
    for (int gg = off >> 6; gg < MAXG; ++gg) group_expert[gg] = -1;
  }
}

__global__ void route_scatter_kernel(const int* __restrict__ expert_of,
                                     const int* __restrict__ offsets,
                                     int* __restrict__ cursor,
                                     int* __restrict__ perm) {
  const int t = blockIdx.x * 256 + threadIdx.x;
  if (t < TT) {
    const int e = expert_of[t];
    const int r = atomicAdd(&cursor[e], 1);
    perm[offsets[e] + r] = t;
  }
}

// ---------------------------------------------------------------------------
// MoE GEMM1 (tiled, double-buffered): h = relu(y1[perm] @ W1[e] + b1[e])
// W1[e]=[D,F] (N contig). 128 thr: M=64 (one group), N=128, KSTEP=32.
// Wave w owns cols [n0+32w, n0+32w+32): 4 M-tiles x 2 N-tiles.
// ---------------------------------------------------------------------------
__global__ __launch_bounds__(128)
void moe_gemm1_tiled(const float* __restrict__ y1,
                     const float* __restrict__ W1,
                     const float* __restrict__ b1,
                     const int* __restrict__ perm,
                     const int* __restrict__ group_expert,
                     __bf16* __restrict__ h) {
  const int g = blockIdx.x;
  const int e = group_expert[g];
  if (e < 0) return;
  __shared__ __align__(16) float As[2][64 * 32];     // [row][k]
  __shared__ __align__(16) float Bs[2][32 * 128];    // [k][col]
  const int tid = threadIdx.x, wave = tid >> 5, lane = tid & 31;
  const int half = lane >> 4, lr = lane & 15;
  const int n0 = blockIdx.y * 128;
  const float* Wb = W1 + (size_t)e * DD * FF;

  v8f acc[4][2] = {};

  auto stage = [&](int kb, int bi) {
#pragma unroll
    for (int i = 0; i < 4; ++i) {                    // A: 512 x 16B chunks
      const int ci = tid + 128 * i;
      const int row = ci >> 3, kc = (ci & 7) * 4;
      const int tok = perm[g * 64 + row];
      if (tok >= 0) async_load16(&As[bi][row * 32 + kc], y1 + (size_t)tok * DD + kb + kc);
      else          *(float4*)&As[bi][row * 32 + kc] = make_float4(0.f, 0.f, 0.f, 0.f);
    }
#pragma unroll
    for (int i = 0; i < 8; ++i) {                    // B: 1024 x 16B chunks
      const int ci = tid + 128 * i;
      const int k = ci >> 5, col = (ci & 31) * 4;
      async_load16(&Bs[bi][k * 128 + col], Wb + (size_t)(kb + k) * FF + n0 + col);
    }
  };

  stage(0, 0);
  wait_async();
  __syncthreads();

  for (int it = 0; it < DD / 32; ++it) {
    const int cur = it & 1;
    if (it + 1 < DD / 32) stage((it + 1) * 32, cur ^ 1);

    v16bf bfr[2];
#pragma unroll
    for (int nt = 0; nt < 2; ++nt) {
      const int col = wave * 32 + nt * 16 + lr;
#pragma unroll
      for (int v = 0; v < 8; ++v) {
        const int kk = b_koff(v, half);
        bfr[nt][2*v]   = (__bf16)Bs[cur][kk * 128 + col];
        bfr[nt][2*v+1] = (__bf16)Bs[cur][(kk + 1) * 128 + col];
      }
    }
#pragma unroll
    for (int mt = 0; mt < 4; ++mt) {
      v16bf afrag;
#pragma unroll
      for (int v = 0; v < 8; ++v) {
        const int ka = a_koff(v, half);
        float2 f2 = *(const float2*)&As[cur][(mt * 16 + lr) * 32 + ka];
        afrag[2*v] = (__bf16)f2.x; afrag[2*v+1] = (__bf16)f2.y;
      }
#pragma unroll
      for (int nt = 0; nt < 2; ++nt)
        acc[mt][nt] = wmma_bf16(afrag, bfr[nt], acc[mt][nt]);
    }
    wait_async();
    __syncthreads();
  }

#pragma unroll
  for (int mt = 0; mt < 4; ++mt) {
#pragma unroll
    for (int nt = 0; nt < 2; ++nt) {
      const int col = n0 + wave * 32 + nt * 16 + lr;
      const float bias = b1[(size_t)e * FF + col];
#pragma unroll
      for (int r = 0; r < 8; ++r) {
        float v = acc[mt][nt][r] + bias;
        v = v > 0.0f ? v : 0.0f;
        h[(size_t)(g * 64 + mt * 16 + r + half * 8) * FF + col] = (__bf16)v;
      }
    }
  }
}

// ---------------------------------------------------------------------------
// MoE GEMM2 (tiled, double-buffered): moe[token] = h @ W2[e] + b2[e]
// ---------------------------------------------------------------------------
__global__ __launch_bounds__(128)
void moe_gemm2_tiled(const __bf16* __restrict__ h,
                     const float* __restrict__ W2,
                     const float* __restrict__ b2,
                     const int* __restrict__ perm,
                     const int* __restrict__ group_expert,
                     float* __restrict__ moe) {
  const int g = blockIdx.x;
  const int e = group_expert[g];
  if (e < 0) return;
  __shared__ __align__(16) __bf16 As[2][64 * 32];    // [row][k]
  __shared__ __align__(16) float  Bs[2][32 * 128];   // [k][col]
  const int tid = threadIdx.x, wave = tid >> 5, lane = tid & 31;
  const int half = lane >> 4, lr = lane & 15;
  const int n0 = blockIdx.y * 128;
  const float* Wb = W2 + (size_t)e * FF * DD;

  v8f acc[4][2] = {};

  auto stage = [&](int kb, int bi) {
#pragma unroll
    for (int i = 0; i < 2; ++i) {                    // A: 256 x 16B chunks
      const int ci = tid + 128 * i;
      const int row = ci >> 2, kc = (ci & 3) * 8;
      async_load16(&As[bi][row * 32 + kc], h + (size_t)(g * 64 + row) * FF + kb + kc);
    }
#pragma unroll
    for (int i = 0; i < 8; ++i) {                    // B: 1024 x 16B chunks
      const int ci = tid + 128 * i;
      const int k = ci >> 5, col = (ci & 31) * 4;
      async_load16(&Bs[bi][k * 128 + col], Wb + (size_t)(kb + k) * DD + n0 + col);
    }
  };

  stage(0, 0);
  wait_async();
  __syncthreads();

  for (int it = 0; it < FF / 32; ++it) {
    const int cur = it & 1;
    if (it + 1 < FF / 32) stage((it + 1) * 32, cur ^ 1);

    v16bf bfr[2];
#pragma unroll
    for (int nt = 0; nt < 2; ++nt) {
      const int col = wave * 32 + nt * 16 + lr;
#pragma unroll
      for (int v = 0; v < 8; ++v) {
        const int kk = b_koff(v, half);
        bfr[nt][2*v]   = (__bf16)Bs[cur][kk * 128 + col];
        bfr[nt][2*v+1] = (__bf16)Bs[cur][(kk + 1) * 128 + col];
      }
    }
#pragma unroll
    for (int mt = 0; mt < 4; ++mt) {
      v16bf afrag;
#pragma unroll
      for (int v = 0; v < 8; ++v) {
        const int ka = a_koff(v, half);
        setpair(afrag, v, *(const unsigned*)&As[cur][(mt * 16 + lr) * 32 + ka]);
      }
#pragma unroll
      for (int nt = 0; nt < 2; ++nt)
        acc[mt][nt] = wmma_bf16(afrag, bfr[nt], acc[mt][nt]);
    }
    wait_async();
    __syncthreads();
  }

#pragma unroll
  for (int mt = 0; mt < 4; ++mt) {
#pragma unroll
    for (int nt = 0; nt < 2; ++nt) {
      const int col = n0 + wave * 32 + nt * 16 + lr;
      const float bias = b2[(size_t)e * DD + col];
#pragma unroll
      for (int r = 0; r < 8; ++r) {
        const int tokr = perm[g * 64 + mt * 16 + r + half * 8];
        if (tokr >= 0) moe[(size_t)tokr * DD + col] = acc[mt][nt][r] + bias;
      }
    }
  }
}

// ---------------------------------------------------------------------------
// Launch
// ---------------------------------------------------------------------------
extern "C" void kernel_launch(void* const* d_in, const int* in_sizes, int n_in,
                              void* d_out, int out_size, void* d_ws, size_t ws_size,
                              hipStream_t stream) {
  const float* x         = (const float*)d_in[0];
  const float* in_proj_w = (const float*)d_in[1];
  const float* in_proj_b = (const float*)d_in[2];
  const float* out_w     = (const float*)d_in[3];
  const float* out_b     = (const float*)d_in[4];
  const float* gate_w    = (const float*)d_in[5];
  const float* gate_b    = (const float*)d_in[6];
  const float* W1        = (const float*)d_in[7];
  const float* b1        = (const float*)d_in[8];
  const float* W2        = (const float*)d_in[9];
  const float* b2        = (const float*)d_in[10];
  const float* ln1_g     = (const float*)d_in[11];
  const float* ln1_b     = (const float*)d_in[12];
  const float* ln2_g     = (const float*)d_in[13];
  const float* ln2_b     = (const float*)d_in[14];
  float* out = (float*)d_out;

  char* ws = (char*)d_ws;
  size_t off = 0;
  auto alloc = [&](size_t bytes) {
    void* p = ws + off;
    off = (off + bytes + 255) & ~(size_t)255;
    return p;
  };
  __bf16* qkv   = (__bf16*)alloc((size_t)TT * N3 * 2);
  __bf16* attn  = (__bf16*)alloc((size_t)TT * DD * 2);
  float*  proj  = (float*) alloc((size_t)TT * DD * 4);
  float*  y1    = (float*) alloc((size_t)TT * DD * 4);
  float*  moe   = (float*) alloc((size_t)TT * DD * 4);
  __bf16* hbuf  = (__bf16*)alloc((size_t)PERMCAP * FF * 2);
  int* expert_of    = (int*)alloc(TT * 4);
  int* counts       = (int*)alloc(EE * 4);
  int* cursor       = (int*)alloc(EE * 4);
  int* offsets      = (int*)alloc((EE + 1) * 4);
  int* group_expert = (int*)alloc(MAXG * 4);
  int* perm         = (int*)alloc(PERMCAP * 4);
  (void)ws_size; (void)n_in; (void)in_sizes; (void)out_size;

  // 1) QKV projection: qkv(bf16) = x @ in_proj_w^T + b
  gemm_dense_tiled<false><<<dim3(TT/128, N3/64), 128, 0, stream>>>(
      x, in_proj_w, in_proj_b, nullptr, qkv, TT, N3, DD);

  // 2) attention (4 query tiles per block)
  attn_kernel<<<dim3(SS/64, HH, BB), 128, 0, stream>>>(qkv, attn);

  // 3) output projection: proj(f32) = attn @ out_w^T + out_b
  gemm_dense_tiled<true><<<dim3(TT/128, DD/64), 128, 0, stream>>>(
      attn, out_w, out_b, proj, nullptr, TT, DD, DD);

  // 4) y1 = LN1(x + proj)
  add_ln_kernel<<<TT, 256, 0, stream>>>(x, proj, ln1_g, ln1_b, y1);

  // 5) routing
  route_init_kernel<<<(PERMCAP + 255)/256, 256, 0, stream>>>(perm, counts, cursor);
  gate_kernel<<<TT/8, dim3(32, 8), 0, stream>>>(y1, gate_w, gate_b, expert_of, counts);
  route_offsets_kernel<<<1, 32, 0, stream>>>(counts, offsets, group_expert);
  route_scatter_kernel<<<(TT + 255)/256, 256, 0, stream>>>(expert_of, offsets, cursor, perm);

  // 6) MoE expert FFN (expert-uniform 64-row groups)
  moe_gemm1_tiled<<<dim3(MAXG, FF/128), 128, 0, stream>>>(
      y1, W1, b1, perm, group_expert, hbuf);
  moe_gemm2_tiled<<<dim3(MAXG, DD/128), 128, 0, stream>>>(
      hbuf, W2, b2, perm, group_expert, moe);

  // 7) out = LN2(y1 + moe)
  add_ln_kernel<<<TT, 256, 0, stream>>>(y1, moe, ln2_g, ln2_b, out);
}
